// RotaryEmbeddingTransformerLayer_54597624267421
// MI455X (gfx1250) — matrix operand order repaired and checked by reading, not compiled, not run on hardware
//
#include <hip/hip_runtime.h>
#include <hip/hip_bf16.h>
#include <math.h>

// ---------------------------------------------------------------------------
// RotaryEmbeddingTransformerLayer for MI455X (gfx1250), bf16 WMMA everywhere.
// B=4, S=2048, d=2048, f=8192. ~1.1 TFLOP of GEMM work on the
// v_wmma_f32_16x16x32_bf16 pipe (wave32).
// GEMMs: double-buffered LDS fed by global_load_async_to_lds_b128 (ASYNCcnt),
// BK=64 -> 16 WMMAs per barrier period. Flash attention streams L2-resident
// bf16 K/V fragments with depth-2 software pipelining.
// ---------------------------------------------------------------------------

typedef __bf16 bf16;
typedef __bf16 v16bf __attribute__((ext_vector_type(16)));
typedef __bf16 v8bf  __attribute__((ext_vector_type(8)));
typedef float  v8f   __attribute__((ext_vector_type(8)));

static __device__ inline v16bf cat8(v8bf lo, v8bf hi) {
  v16bf r;
#pragma unroll
  for (int i = 0; i < 8; ++i) { r[i] = lo[i]; r[i + 8] = hi[i]; }
  return r;
}

// async copy 16B cache->LDS, tracked by ASYNCcnt (CDNA5 §15.18 opcode 98)
#define ASYNC_LDS_B128(ldsoff, gptr)                                   \
  asm volatile("global_load_async_to_lds_b128 %0, %1, off"             \
               :: "v"(ldsoff), "v"(gptr) : "memory")
#define WAIT_ASYNC0() asm volatile("s_wait_asynccnt 0x0" ::: "memory")

// ---------------------------------------------------------------------------
// f32 -> bf16 elementwise conversion (weights)
// ---------------------------------------------------------------------------
__global__ __launch_bounds__(256) void k_cvt(const float* __restrict__ x,
                                             bf16* __restrict__ y, size_t n) {
  size_t i = (size_t)blockIdx.x * 256 + threadIdx.x;
  if (i < n) y[i] = (bf16)x[i];
}

// ---------------------------------------------------------------------------
// LayerNorm over last dim (d). One 256-thread block per row.
// ---------------------------------------------------------------------------
__global__ __launch_bounds__(256)
void k_layernorm(const float* __restrict__ x, const float* __restrict__ g,
                 const float* __restrict__ b, float* __restrict__ yf,
                 bf16* __restrict__ yb, int d) {
  __shared__ float red[256];
  int row = blockIdx.x, tid = threadIdx.x;
  const float* xr = x + (size_t)row * d;
  float s = 0.f;
  for (int i = tid; i < d; i += 256) s += xr[i];
  red[tid] = s; __syncthreads();
  for (int o = 128; o > 0; o >>= 1) {
    if (tid < o) red[tid] += red[tid + o];
    __syncthreads();
  }
  float mean = red[0] / d;
  __syncthreads();
  float v = 0.f;
  for (int i = tid; i < d; i += 256) { float t = xr[i] - mean; v += t * t; }
  red[tid] = v; __syncthreads();
  for (int o = 128; o > 0; o >>= 1) {
    if (tid < o) red[tid] += red[tid + o];
    __syncthreads();
  }
  float rstd = rsqrtf(red[0] / d + 1e-5f);
  for (int i = tid; i < d; i += 256) {
    float y = (xr[i] - mean) * rstd * g[i] + b[i];
    if (yf) yf[(size_t)row * d + i] = y;
    yb[(size_t)row * d + i] = (bf16)y;
  }
}

// ---------------------------------------------------------------------------
// Tiled bf16 WMMA GEMM:  C[M,N] = epi(A[M,K] @ W[N,K]^T + bias[N])
// 128x128 block tile, BK=64, 8 waves each owning a 32x64 sub-tile.
// Double-buffered dynamic LDS (2 x 16KB per matrix) filled by
// global_load_async_to_lds_b128; one s_wait_asynccnt + barrier per K-step.
// Fragment layouts per CDNA5 ISA 7.12.2:
//   A 16x32: lane l<16 holds row l, K {0..7,16..23}; lane l+16: K {8..15,24..31}
//   B 32x16: lane n<16 holds col n, K {0..7,16..23}; lane n+16: K {8..15,24..31}
// ---------------------------------------------------------------------------
#define BM 128
#define BN 128
#define BK 64

template <bool RELU, bool RESID, bool OUTBF16>
__global__ __launch_bounds__(256)
void k_gemm_bf16(const bf16* __restrict__ A, const bf16* __restrict__ W,
                 const float* __restrict__ bias, const float* __restrict__ resid,
                 float* __restrict__ Cf, bf16* __restrict__ Cb,
                 int M, int N, int K) {
  extern __shared__ bf16 smem[];
  bf16* sA = smem;                 // [2][BM][BK]
  bf16* sB = smem + 2 * BM * BK;   // [2][BN][BK]
  unsigned ldsbase = __builtin_amdgcn_groupstaticsize();
  unsigned offA = ldsbase;
  unsigned offB = ldsbase + 2u * BM * BK * 2u;

  int tid = threadIdx.x;
  int wave = tid >> 5, lane = tid & 31;
  int half = lane >> 4, l16 = lane & 15;
  int wm = (wave >> 1) * 32;   // wave row offset inside block tile
  int wn = (wave & 1) * 64;    // wave col offset inside block tile
  int m0 = blockIdx.y * BM;
  int n0 = blockIdx.x * BN;

  v8f acc[2][4] = {};

  // issue one 128x64 bf16 tile (A and B) into LDS buffer `buf`
  auto issue_tile = [&](int buf, int kblk) {
#pragma unroll
    for (int t = 0; t < 4; ++t) {
      int idx = (tid + t * 256) * 8;     // element index within tile
      int r = idx >> 6, c = idx & 63;
      unsigned la = offA + (unsigned)((buf * BM + r) * BK + c) * 2u;
      unsigned lb = offB + (unsigned)((buf * BN + r) * BK + c) * 2u;
      const bf16* ga = &A[(size_t)(m0 + r) * K + (size_t)kblk * BK + c];
      const bf16* gb = &W[(size_t)(n0 + r) * K + (size_t)kblk * BK + c];
      ASYNC_LDS_B128(la, ga);
      ASYNC_LDS_B128(lb, gb);
    }
  };

  auto compute = [&](int buf) {
#pragma unroll
    for (int ks = 0; ks < 2; ++ks) {
      v16bf aF[2], bF[4];
#pragma unroll
      for (int r = 0; r < 2; ++r) {
        const bf16* p =
            sA + (size_t)(buf * BM + wm + r * 16 + l16) * BK + ks * 32 + half * 8;
        aF[r] = cat8(*(const v8bf*)p, *(const v8bf*)(p + 16));
      }
#pragma unroll
      for (int c = 0; c < 4; ++c) {
        const bf16* p =
            sB + (size_t)(buf * BN + wn + c * 16 + l16) * BK + ks * 32 + half * 8;
        bF[c] = cat8(*(const v8bf*)p, *(const v8bf*)(p + 16));
      }
#pragma unroll
      for (int r = 0; r < 2; ++r)
#pragma unroll
        for (int c = 0; c < 4; ++c)
          acc[r][c] = __builtin_amdgcn_wmma_f32_16x16x32_bf16(
              false, aF[r], false, bF[c], (short)0, acc[r][c], false, false);
    }
  };

  // prologue: tile 0 -> LDS[0]
  issue_tile(0, 0);
  WAIT_ASYNC0();
  __syncthreads();

  int nsteps = K / BK;
  for (int i = 0; i < nsteps; ++i) {
    int cur = i & 1, nxt = cur ^ 1;
    bool more = (i + 1 < nsteps);
    if (more) issue_tile(nxt, i + 1);   // async fill of other buffer
    compute(cur);                        // 16 WMMAs from current buffer
    if (more) {
      WAIT_ASYNC0();                     // my async writes to nxt landed
      __syncthreads();                   // everyone's landed; cur free to refill
    }
  }

  // epilogue: C/D layout -> VGPR v holds row (v + 8*half), col = l16
#pragma unroll
  for (int r = 0; r < 2; ++r) {
#pragma unroll
    for (int c = 0; c < 4; ++c) {
#pragma unroll
      for (int v = 0; v < 8; ++v) {
        int row = m0 + wm + r * 16 + v + half * 8;
        int col = n0 + wn + c * 16 + l16;
        size_t idx = (size_t)row * N + col;
        float x = acc[r][c][v] + bias[col];
        if (RESID) x += resid[idx];
        if (RELU)  x = fmaxf(x, 0.f);
        if (OUTBF16) Cb[idx] = (bf16)x; else Cf[idx] = x;
      }
    }
  }
}

// ---------------------------------------------------------------------------
// RoPE exactly as reference (incl. cos(emb)/sin(emb) of the sinusoid table):
//   emb[p][j] = j<h ? sin(p*invf[j]) : cos(p*invf[j-h]);  c=cos(emb), s=sin(emb)
//   rot(x)[j] = j<h ? -x[2j] : x[2(j-h)+1]
//   Q = Q*c + rot(Q)*s ;  K = K*c - rot(K)*s
// ---------------------------------------------------------------------------
__global__ __launch_bounds__(256)
void k_rope(const bf16* __restrict__ qkv, bf16* __restrict__ Qo,
            bf16* __restrict__ Ko, int S, int d) {
  size_t i = (size_t)blockIdx.x * 256 + threadIdx.x;  // over T*d
  int j = (int)(i % d);
  size_t tok = i / d;
  int pos = (int)(tok % S);
  int h = d >> 1;
  int fi = (j < h) ? j : (j - h);
  float invf = __powf(10000.f, -((float)(2 * fi) / (float)d));
  float sp = (float)pos * invf;
  float emb = (j < h) ? __sinf(sp) : __cosf(sp);
  float c = __cosf(emb), s = __sinf(emb);
  const bf16* row = qkv + tok * (size_t)(3 * d);
  float qj = (float)row[j];
  float kj = (float)row[d + j];
  float qr, kr;
  if (j < h) { qr = -(float)row[2 * j];           kr = -(float)row[d + 2 * j]; }
  else       { qr =  (float)row[2 * (j - h) + 1]; kr =  (float)row[d + 2 * (j - h) + 1]; }
  Qo[i] = (bf16)(qj * c + qr * s);
  Ko[i] = (bf16)(kj * c - kr * s);
}

// ---------------------------------------------------------------------------
// V transpose, 32x32 LDS-tiled (coalesced both sides):
//   Vt[b][j][s] = qkv[(b*S+s)*3d + 2d + j]
// ---------------------------------------------------------------------------
__global__ __launch_bounds__(256)
void k_vtrans(const bf16* __restrict__ qkv, bf16* __restrict__ Vt, int S, int d) {
  __shared__ bf16 tile[32][33];
  int j0 = blockIdx.x * 32;          // feature index
  int s0 = blockIdx.y * 32;          // sequence index
  int b  = blockIdx.z;
  int tx = threadIdx.x & 31, ty = threadIdx.x >> 5;   // 32 x 8
#pragma unroll
  for (int r = 0; r < 32; r += 8)
    tile[ty + r][tx] =
        qkv[((size_t)b * S + s0 + ty + r) * (size_t)(3 * d) + 2 * d + j0 + tx];
  __syncthreads();
#pragma unroll
  for (int r = 0; r < 32; r += 8)
    Vt[((size_t)b * d + j0 + ty + r) * S + s0 + tx] = tile[tx][ty + r];
}

// ---------------------------------------------------------------------------
// Flash attention, causal, single head of dim d=2048.
// One block (8 waves) per 16-query tile. Wave w owns d-slice [256w,256w+256):
//  - Q fragments preloaded (8 x v16bf), O accumulator 16 x v8f.
// Per 32-key block: QK^T partials -> ds_add_f32 reduce into LDS, online
// softmax parallel over 32 lanes (shfl_xor cross-half reduce), rescale O,
// PV via WMMA. K/V fragment loads are depth-2 software pipelined.
// ---------------------------------------------------------------------------
__global__ __launch_bounds__(256)
void k_flash(const bf16* __restrict__ Q, const bf16* __restrict__ Km,
             const bf16* __restrict__ Vt, bf16* __restrict__ O,
             int S, int d, float scale) {
  int bq = blockIdx.x;
  int b = bq / (S / 16);
  int q0 = (bq % (S / 16)) * 16;
  const bf16* Qp = Q + ((size_t)b * S + q0) * d;
  const bf16* Kp = Km + (size_t)b * S * d;
  const bf16* Vp = Vt + (size_t)b * d * S;   // [d][S]

  int tid = threadIdx.x;
  int wave = tid >> 5, lane = tid & 31;
  int half = lane >> 4, l16 = lane & 15;
  int d0 = wave * 256;                       // this wave's d / output-col slice

  // preload Q fragments (A operand, 16x32 each) for this wave's slice
  v16bf qF[8];
#pragma unroll
  for (int s = 0; s < 8; ++s) {
    const bf16* p = Qp + (size_t)l16 * d + d0 + s * 32 + half * 8;
    qF[s] = cat8(*(const v8bf*)p, *(const v8bf*)(p + 16));
  }

  v8f oacc[16] = {};

  __shared__ float sS[16][32];
  __shared__ bf16  sP[16][32];
  __shared__ float sm[16], sl[16], salpha[16];
  if (tid < 16) { sm[tid] = -1e30f; sl[tid] = 0.f; }

  int nblocks = (q0 + 16 + 31) / 32;
  for (int j = 0; j < nblocks; ++j) {
    int kv0 = j * 32;
    float* sSf = &sS[0][0];
    sSf[tid] = 0.f; sSf[tid + 256] = 0.f;
    __syncthreads();

    // ---- S_partial = Q_slice @ K_slice^T, depth-2 pipelined B-frag loads ----
    v8f sacc[2] = {};
    auto kload = [&](int it) -> v16bf {
      int s = it >> 1, nt = it & 1;
      const bf16* p =
          Kp + (size_t)(kv0 + nt * 16 + l16) * d + d0 + s * 32 + half * 8;
      return cat8(*(const v8bf*)p, *(const v8bf*)(p + 16));
    };
    v16bf kf0 = kload(0);
    v16bf kf1 = kload(1);
#pragma unroll
    for (int it = 0; it < 16; ++it) {
      v16bf kn = kf0;
      if (it + 2 < 16) kn = kload(it + 2);
      sacc[it & 1] = __builtin_amdgcn_wmma_f32_16x16x32_bf16(
          false, qF[it >> 1], false, kf0, (short)0, sacc[it & 1], false, false);
      kf0 = kf1; kf1 = kn;
    }
    // cross-wave reduction into LDS (ds_add_f32)
#pragma unroll
    for (int nt = 0; nt < 2; ++nt)
#pragma unroll
      for (int v = 0; v < 8; ++v)
        atomicAdd(&sS[v + half * 8][nt * 16 + l16], sacc[nt][v]);
    __syncthreads();

    // ---- online softmax with causal mask (32 lanes: lane = row + 16*colhalf)
    if (tid < 32) {
      int r = tid & 15, q = q0 + r;
      int c0 = (tid >> 4) * 16;
      float xs[16];
      float mloc = -1e30f;
#pragma unroll
      for (int c = 0; c < 16; ++c) {
        int key = kv0 + c0 + c;
        float x = (key <= q) ? sS[r][c0 + c] * scale : -1e30f;
        xs[c] = x;
        mloc = fmaxf(mloc, x);
      }
      float mold = sm[r];
      float mnew = fmaxf(mold, fmaxf(mloc, __shfl_xor(mloc, 16)));
      float lsum = 0.f;
#pragma unroll
      for (int c = 0; c < 16; ++c) {
        float p = __expf(xs[c] - mnew);
        sP[r][c0 + c] = (bf16)p;
        lsum += p;
      }
      lsum += __shfl_xor(lsum, 16);
      if (tid < 16) {
        float alpha = __expf(mold - mnew);
        sm[r] = mnew;
        sl[r] = sl[r] * alpha + lsum;
        salpha[r] = alpha;
      }
    }
    __syncthreads();

    // ---- rescale accumulators ----
    float av[8];
#pragma unroll
    for (int v = 0; v < 8; ++v) av[v] = salpha[v + half * 8];
#pragma unroll
    for (int t = 0; t < 16; ++t)
#pragma unroll
      for (int v = 0; v < 8; ++v) oacc[t][v] *= av[v];

    // ---- O_slice += P @ V_block, depth-2 pipelined V-frag loads ----
    const bf16* pp = &sP[l16][half * 8];
    v16bf pF = cat8(*(const v8bf*)pp, *(const v8bf*)(pp + 16));
    auto vload = [&](int t) -> v16bf {
      const bf16* vp = Vp + (size_t)(d0 + t * 16 + l16) * S + kv0 + half * 8;
      return cat8(*(const v8bf*)vp, *(const v8bf*)(vp + 16));
    };
    v16bf vf0 = vload(0);
    v16bf vf1 = vload(1);
#pragma unroll
    for (int t = 0; t < 16; ++t) {
      v16bf vn = vf0;
      if (t + 2 < 16) vn = vload(t + 2);
      oacc[t] = __builtin_amdgcn_wmma_f32_16x16x32_bf16(
          false, pF, false, vf0, (short)0, oacc[t], false, false);
      vf0 = vf1; vf1 = vn;
    }
    __syncthreads();
  }

  // normalize and store bf16
  float lv[8];
#pragma unroll
  for (int v = 0; v < 8; ++v) lv[v] = 1.f / sl[v + half * 8];
#pragma unroll
  for (int t = 0; t < 16; ++t) {
    int col = d0 + t * 16 + l16;
#pragma unroll
    for (int v = 0; v < 8; ++v) {
      int row = q0 + v + half * 8;
      O[((size_t)b * S + row) * d + col] = (bf16)(oacc[t][v] * lv[v]);
    }
  }
}

// ---------------------------------------------------------------------------
// Host-side orchestration
// ---------------------------------------------------------------------------
extern "C" void kernel_launch(void* const* d_in, const int* in_sizes, int n_in,
                              void* d_out, int out_size, void* d_ws, size_t ws_size,
                              hipStream_t stream) {
  const int B = 4, S = 2048, d = 2048, f = 8192;
  const int T = B * S, D3 = 3 * d;
  const size_t GEMM_LDS = 2u * (BM + BN) * BK * 2u;  // 64KB dynamic

  const float* src  = (const float*)d_in[0];
  const float* Wqkv = (const float*)d_in[1];
  const float* bqkv = (const float*)d_in[2];
  const float* Wo   = (const float*)d_in[3];
  const float* bo   = (const float*)d_in[4];
  const float* W1   = (const float*)d_in[5];
  const float* b1   = (const float*)d_in[6];
  const float* W2   = (const float*)d_in[7];
  const float* b2   = (const float*)d_in[8];
  const float* g1   = (const float*)d_in[9];
  const float* be1  = (const float*)d_in[10];
  const float* g2   = (const float*)d_in[11];
  const float* be2  = (const float*)d_in[12];
  float* out = (float*)d_out;

  // workspace carving (256B aligned)
  char* w = (char*)d_ws;
  auto carve = [&](size_t bytes) -> char* {
    char* p = w;
    w += (bytes + 255) & ~(size_t)255;
    return p;
  };
  float* ln1f  = (float*)carve((size_t)T * d * 4);
  bf16* xb     = (bf16*) carve((size_t)T * d * 2);
  bf16* Wqkvb  = (bf16*) carve((size_t)D3 * d * 2);
  bf16* Wob    = (bf16*) carve((size_t)d * d * 2);
  bf16* W1b    = (bf16*) carve((size_t)f * d * 2);
  bf16* W2b    = (bf16*) carve((size_t)d * f * 2);
  bf16* qkvb   = (bf16*) carve((size_t)T * D3 * 2);
  bf16* Qb     = (bf16*) carve((size_t)T * d * 2);
  bf16* Kb     = (bf16*) carve((size_t)T * d * 2);
  bf16* Vtb    = (bf16*) carve((size_t)B * d * S * 2);
  bf16* Ob     = (bf16*) carve((size_t)T * d * 2);
  bf16* ln2b   = (bf16*) carve((size_t)T * d * 2);
  bf16* h1b    = (bf16*) carve((size_t)T * f * 2);

  // 1. weights -> bf16
  k_cvt<<<(int)(((size_t)D3 * d + 255) / 256), 256, 0, stream>>>(Wqkv, Wqkvb, (size_t)D3 * d);
  k_cvt<<<(int)(((size_t)d * d + 255) / 256), 256, 0, stream>>>(Wo, Wob, (size_t)d * d);
  k_cvt<<<(int)(((size_t)f * d + 255) / 256), 256, 0, stream>>>(W1, W1b, (size_t)f * d);
  k_cvt<<<(int)(((size_t)d * f + 255) / 256), 256, 0, stream>>>(W2, W2b, (size_t)d * f);

  // 2. LN1 (f32 residual copy + bf16 GEMM input)
  k_layernorm<<<T, 256, 0, stream>>>(src, g1, be1, ln1f, xb, d);

  // 3. QKV GEMM: qkv = ln1 @ Wqkv^T + bqkv  (bf16 out)
  {
    dim3 g(D3 / BN, T / BM);
    k_gemm_bf16<false, false, true><<<g, 256, GEMM_LDS, stream>>>(
        xb, Wqkvb, bqkv, nullptr, nullptr, qkvb, T, D3, d);
  }

  // 4. RoPE -> Qb, Kb ; 5. V transpose -> Vtb
  k_rope<<<(int)(((size_t)T * d) / 256), 256, 0, stream>>>(qkvb, Qb, Kb, S, d);
  {
    dim3 g(d / 32, S / 32, B);
    k_vtrans<<<g, 256, 0, stream>>>(qkvb, Vtb, S, d);
  }

  // 6. flash attention
  float scale = 1.0f / sqrtf((float)d);
  k_flash<<<B * (S / 16), 256, 0, stream>>>(Qb, Kb, Vtb, Ob, S, d, scale);

  // 7. Wo GEMM + residual(ln1) -> d_out (f32): src_after = ln1 + attn@Wo^T + bo
  {
    dim3 g(d / BN, T / BM);
    k_gemm_bf16<false, true, false><<<g, 256, GEMM_LDS, stream>>>(
        Ob, Wob, bo, ln1f, out, nullptr, T, d, d);
  }

  // 8. LN2 -> bf16
  k_layernorm<<<T, 256, 0, stream>>>(out, g2, be2, nullptr, ln2b, d);

  // 9. FFN1: h = relu(ln2 @ W1^T + b1) (bf16 out)
  {
    dim3 g(f / BN, T / BM);
    k_gemm_bf16<true, false, true><<<g, 256, GEMM_LDS, stream>>>(
        ln2b, W1b, b1, nullptr, nullptr, h1b, T, f, d);
  }

  // 10. FFN2 + residual(src_after): out = src_after + h @ W2^T + b2
  {
    dim3 g(d / BN, T / BM);
    k_gemm_bf16<false, true, false><<<g, 256, GEMM_LDS, stream>>>(
        h1b, W2b, b2, out, out, nullptr, T, d, f);
  }
}
